// TransformerDecoder_53695681134868
// MI455X (gfx1250) — matrix operand order repaired and checked by reading, not compiled, not run on hardware
//
#include <hip/hip_runtime.h>
#include <math.h>

// ---------------- model dims ----------------
#define TQ   64
#define NB   8
#define DD   256
#define HH   8
#define HDD  32
#define DFFN 1024
#define NCC  128
#define SSRC 256

typedef __attribute__((ext_vector_type(16))) _Float16     v16h;
typedef __attribute__((ext_vector_type(8)))  float        v8f;
typedef __attribute__((ext_vector_type(4)))  unsigned int u32x4;

union FragU { v16h v; u32x4 u[2]; };

// ---------------- workspace layout ----------------
// f16 region (offsets in halves)
#define OFF_EMBW   0u
#define OFF_SAQKV  32768u
#define OFF_SAWO   425984u
#define OFF_CAWQ   557056u
#define OFF_CAWKV  688128u
#define OFF_CAWO   950272u
#define OFF_FFW1   1081344u
#define OFF_FFW2   1605632u
#define OFF_OPW1   2129920u
#define OFF_OPW2   2195456u
#define OFF_ENC    2228224u
#define OFF_X16    2752512u   // [16][256] padded activation (rows 8..15 stay zero)
#define OFF_ATT16  2756608u   // [16][256]
#define OFF_FFN16  2760704u   // [16][1024]
#define OFF_HEAD16 2777088u   // [16][256]
#define OFF_CUR16  2781184u   // [16][128]
#define F16_END    2783232u
#define F32_BYTE_OFF (F16_END * 2u)
// f32 region (offsets in floats)
#define OFF_MEMK 0u          // [2][8][8][256][32]
#define OFF_MEMV 1048576u
#define OFF_KC   2097152u    // [2][8][64][256]
#define OFF_VC   2359296u
#define OFF_X32  2621440u    // [8][256]
#define OFF_Q32  2623488u    // [8][256]
#define OFF_PROJ 2625536u    // [8][256]

// ---------------- gemm epilogue modes ----------------
#define GM_PLAIN  0
#define GM_EMBED  1
#define GM_QKV    2
#define GM_RELU   3
#define GM_HEADBN 4
#define GM_OUT    5

// ---------------- helpers ----------------
__device__ __forceinline__ float pe_val(int t, int n) {
  float freq = expf(-(float)(n & ~1) * (9.210340371976184f / 256.0f));
  float a = (float)t * freq;
  return (n & 1) ? cosf(a) : sinf(a);
}

// One-wave 16x16 output tile: C = A[16,K] * W_rows[n..n+15, K]^T, f16 in / f32 acc.
__device__ __forceinline__ v8f wmma_tile(const _Float16* __restrict__ A, int lda,
                                         const _Float16* __restrict__ wrow,
                                         int K, int lane) {
  const int mrow = lane & 15;
  const int aoff = (lane & 16) ? 8 : 0;    // A frag: K-halves {0..7,16..23} / {8..15,24..31}
  const int boff = (lane & 16) ? 8 : 0;    // B frag: K-halves {0..7,16..23} / {8..15,24..31}
  const _Float16* ar = A + (size_t)mrow * lda;
  v8f c = {0.f, 0.f, 0.f, 0.f, 0.f, 0.f, 0.f, 0.f};
  for (int kk = 0; kk < K; kk += 32) {
    if (kk + 64 < K) __builtin_prefetch(wrow + kk + 64, 0, 3);  // WGP-scope prefetch
    FragU fa, fb;
    fa.u[0] = *(const u32x4*)(ar + kk + aoff);
    fa.u[1] = *(const u32x4*)(ar + kk + aoff + 16);
    fb.u[0] = *(const u32x4*)(wrow + kk + boff);
    fb.u[1] = *(const u32x4*)(wrow + kk + boff + 16);
    c = __builtin_amdgcn_wmma_f32_16x16x32_f16(false, fa.v, false, fb.v,
                                               (short)0, c, false, false);
  }
  return c;
}

// ---------------- setup kernels ----------------
__global__ void cvt_f32_f16(const float* __restrict__ s, _Float16* __restrict__ d, int n) {
  int i = blockIdx.x * blockDim.x + threadIdx.x;
  if (i < n) d[i] = (_Float16)s[i];
}

__global__ void zero_f16(_Float16* d, int n) {
  int i = blockIdx.x * blockDim.x + threadIdx.x;
  if (i < n) d[i] = (_Float16)0.0f;
}

// Cross-attn K/V precompute (full-GPU WMMA GEMM):
// [L,B,S,512] = enc[B,S,256] @ ca_wkv[L,512,256]^T -> mem_k/mem_v [L,B,H,S,HD] fp32.
__global__ __launch_bounds__(32)
void memkv_gemm(const _Float16* __restrict__ enc16, const _Float16* __restrict__ wkv16,
                const float* __restrict__ ca_bqkv,
                float* __restrict__ mem_k, float* __restrict__ mem_v) {
  const int bid = blockIdx.x;
  const int l = bid >> 12, rem = bid & 4095;
  const int mt = rem >> 5, nt = rem & 31;
  const int lane = threadIdx.x;
  const _Float16* A = enc16 + (size_t)mt * 16 * 256;
  const int n = nt * 16 + (lane & 15);
  const _Float16* wrow = wkv16 + (size_t)l * 512 * 256 + (size_t)n * 256;
  v8f c = wmma_tile(A, 256, wrow, 256, lane);
  const float bias = ca_bqkv[l * 768 + 256 + n];
  const int h = (n & 255) >> 5, hd = n & 31;
  float* dst = (n < 256) ? mem_k : mem_v;
  #pragma unroll
  for (int i = 0; i < 8; ++i) {
    int m = ((lane >> 4) << 3) + i;
    int gr = mt * 16 + m;
    int b = gr >> 8, s = gr & 255;
    dst[((((size_t)l * 8 + b) * 8 + h) * 256 + s) * 32 + hd] = c[i] + bias;
  }
}

// ---------------- persistent decode ----------------
struct DecodeParams {
  const _Float16 *embw, *saqkv, *sawo, *cawq, *cawo, *ffw1, *ffw2, *opw1, *opw2;
  _Float16 *x16, *att16, *ffn16, *head16, *cur16;
  const float *emb_b, *sa_bqkv, *sa_bo, *ca_bqkv, *ca_bo;
  const float *ln1_g, *ln1_b, *ln2_g, *ln2_b, *ln3_g, *ln3_b;
  const float *ffn_b1, *ffn_b2, *op_b1, *bn_g, *bn_b, *op_b2;
  const float *mem_k, *mem_v;
  float *kc, *vc, *x32, *q32, *proj;
  float *dout;
};

// One 16x16 tile of Y[8,N] = A16[16,K] @ W[N,K]^T with fused epilogue.
__device__ __forceinline__ void gemm_tile_step(
    const _Float16* __restrict__ A, const _Float16* __restrict__ W,
    const float* __restrict__ bias, int K, int mode, int t, int ldo, int tile, int lane,
    float* __restrict__ of32, _Float16* __restrict__ of16,
    float* __restrict__ kc, float* __restrict__ vc,
    const float* __restrict__ bng, const float* __restrict__ bnb,
    float* __restrict__ dout) {
  const int n = tile * 16 + (lane & 15);
  v8f c = wmma_tile(A, K, W + (size_t)n * K, K, lane);
  if (lane < 16) {   // rows 0..7 hold the 8 real tokens
    const float bn = bias[n];
    #pragma unroll
    for (int m = 0; m < 8; ++m) {
      float v = c[m] + bn;
      switch (mode) {
        case GM_PLAIN:
          of32[m * ldo + n] = v;
          break;
        case GM_EMBED:
          v += pe_val(t, n);
          of32[m * ldo + n] = v;
          of16[m * ldo + n] = (_Float16)v;
          break;
        case GM_QKV:
          if (n < 256)      of32[m * 256 + n] = v;                        // q
          else if (n < 512) kc[((size_t)m * TQ + t) * 256 + (n - 256)] = v;
          else              vc[((size_t)m * TQ + t) * 256 + (n - 512)] = v;
          break;
        case GM_RELU:
          of16[m * ldo + n] = (_Float16)fmaxf(v, 0.0f);
          break;
        case GM_HEADBN:
          v = fmaxf(v * 0.9999950000374997f * bng[n] + bnb[n], 0.0f);
          of16[m * ldo + n] = (_Float16)v;
          break;
        case GM_OUT:
          dout[((size_t)m * TQ + t) * NCC + n] = v;
          of16[m * NCC + n] = (_Float16)v;   // next step's decoder input
          break;
      }
    }
  }
}

// Wave-local LayerNorm of one batch row (shuffle reductions only, no barriers).
__device__ __forceinline__ void ln_wave(int r, int lane, float* __restrict__ x,
                                        const float* __restrict__ res,
                                        const float* __restrict__ g,
                                        const float* __restrict__ bt,
                                        _Float16* __restrict__ x16) {
  float v[8];
  float sum = 0.f;
  #pragma unroll
  for (int i = 0; i < 8; ++i) {
    v[i] = x[r * 256 + lane + 32 * i] + res[r * 256 + lane + 32 * i];
    sum += v[i];
  }
  for (int o = 16; o > 0; o >>= 1) sum += __shfl_xor(sum, o, 32);
  float mean = sum * (1.f / 256.f);
  float vs = 0.f;
  #pragma unroll
  for (int i = 0; i < 8; ++i) { v[i] -= mean; vs += v[i] * v[i]; }
  for (int o = 16; o > 0; o >>= 1) vs += __shfl_xor(vs, o, 32);
  float inv = rsqrtf(vs * (1.f / 256.f) + 1e-5f);
  #pragma unroll
  for (int i = 0; i < 8; ++i) {
    int n = lane + 32 * i;
    float o = v[i] * inv * g[n] + bt[n];
    x[r * 256 + n] = o;
    x16[r * 256 + n] = (_Float16)o;
  }
}

// Incremental causal self-attention, one (b,h) task per call. 3 uniform barriers inside.
__device__ __forceinline__ void self_attn_task(const DecodeParams& p, int l, int task,
                                               int lane, int t,
                                               float* __restrict__ ssw,
                                               float* __restrict__ sqw) {
  const float* kcL = p.kc + (size_t)l * 131072;
  const float* vcL = p.vc + (size_t)l * 131072;
  const int b = task >> 3, hh = task & 7;
  sqw[lane] = p.q32[b * 256 + hh * 32 + lane];
  __syncthreads();
  const float scale = 0.17677669529663687f;  // 1/sqrt(32)
  for (int jb = 0; jb < TQ; jb += 32) {
    int j = jb + lane;
    float s = -1e30f;
    if (j <= t) {
      const float* kp = kcL + ((size_t)b * TQ + j) * 256 + hh * 32;
      float acc = 0.f;
      #pragma unroll
      for (int d = 0; d < 32; ++d) acc += sqw[d] * kp[d];
      s = acc * scale;
    }
    ssw[j] = s;
  }
  __syncthreads();
  float v0 = ssw[lane], v1 = ssw[lane + 32];
  float m = fmaxf(v0, v1);
  for (int o = 16; o > 0; o >>= 1) m = fmaxf(m, __shfl_xor(m, o, 32));
  float e0 = (v0 > -1e29f) ? expf(v0 - m) : 0.f;
  float e1 = (v1 > -1e29f) ? expf(v1 - m) : 0.f;
  float sum = e0 + e1;
  for (int o = 16; o > 0; o >>= 1) sum += __shfl_xor(sum, o, 32);
  float inv = 1.f / sum;
  ssw[lane] = e0 * inv;
  ssw[lane + 32] = e1 * inv;
  __syncthreads();
  float acc = 0.f;
  for (int j = 0; j <= t; ++j)
    acc += ssw[j] * vcL[((size_t)b * TQ + j) * 256 + hh * 32 + lane];
  p.att16[b * 256 + hh * 32 + lane] = (_Float16)acc;
}

// Cross-attention over 256 encoder positions, one (b,h) task. 3 uniform barriers inside.
__device__ __forceinline__ void cross_attn_task(const DecodeParams& p, int l, int task,
                                                int lane,
                                                float* __restrict__ ssw,
                                                float* __restrict__ sqw) {
  const int b = task >> 3, hh = task & 7;
  const float* mk = p.mem_k + (size_t)l * 524288 + (size_t)(b * 8 + hh) * SSRC * 32;
  const float* mv = p.mem_v + (size_t)l * 524288 + (size_t)(b * 8 + hh) * SSRC * 32;
  sqw[lane] = p.q32[b * 256 + hh * 32 + lane];
  __syncthreads();
  const float scale = 0.17677669529663687f;
  for (int jb = 0; jb < SSRC; jb += 32) {
    const float* kp = mk + (size_t)(jb + lane) * 32;
    float acc = 0.f;
    #pragma unroll
    for (int d = 0; d < 32; ++d) acc += sqw[d] * kp[d];
    ssw[jb + lane] = acc * scale;
  }
  __syncthreads();
  float vreg[8], e[8];
  float m = -1e30f;
  #pragma unroll
  for (int i = 0; i < 8; ++i) { vreg[i] = ssw[lane + 32 * i]; m = fmaxf(m, vreg[i]); }
  for (int o = 16; o > 0; o >>= 1) m = fmaxf(m, __shfl_xor(m, o, 32));
  float sum = 0.f;
  #pragma unroll
  for (int i = 0; i < 8; ++i) { e[i] = expf(vreg[i] - m); sum += e[i]; }
  for (int o = 16; o > 0; o >>= 1) sum += __shfl_xor(sum, o, 32);
  float inv = 1.f / sum;
  #pragma unroll
  for (int i = 0; i < 8; ++i) ssw[lane + 32 * i] = e[i] * inv;
  __syncthreads();
  float acc = 0.f;
  for (int j = 0; j < SSRC; ++j) acc += ssw[j] * mv[(size_t)j * 32 + lane];
  p.att16[b * 256 + hh * 32 + lane] = (_Float16)acc;
}

// The entire 64-step autoregressive decode in ONE launch: 16 wave32s on one WGP,
// workgroup barriers between stages, WMMA for every dense matmul.
__global__ __launch_bounds__(512, 1)
void decode_persistent(DecodeParams p) {
  const int tid = threadIdx.x;
  const int w = tid >> 5, lane = tid & 31;
  __shared__ float ss[16][SSRC];   // per-wave attention score scratch (16 KB)
  __shared__ float sq[16][HDD];    // per-wave q scratch (2 KB)
  float* ssw = ss[w];
  float* sqw = sq[w];

  for (int t = 0; t < TQ; ++t) {
    // embed + positional encoding: 16 tiles, one per wave
    gemm_tile_step(p.cur16, p.embw, p.emb_b, 128, GM_EMBED, t, 256, w, lane,
                   p.x32, p.x16, nullptr, nullptr, nullptr, nullptr, nullptr);
    __syncthreads();

    for (int l = 0; l < 2; ++l) {
      float* kcL = p.kc + (size_t)l * 131072;
      float* vcL = p.vc + (size_t)l * 131072;

      // QKV projection: 48 tiles, 3 per wave (uniform trip count)
      for (int tile = w; tile < 48; tile += 16)
        gemm_tile_step(p.x16, p.saqkv + (size_t)l * 196608, p.sa_bqkv + l * 768,
                       256, GM_QKV, t, 0, tile, lane,
                       p.q32, nullptr, kcL, vcL, nullptr, nullptr, nullptr);
      __syncthreads();

      // causal self-attention: 64 (b,h) tasks, 4 per wave (uniform)
      for (int task = w; task < 64; task += 16)
        self_attn_task(p, l, task, lane, t, ssw, sqw);
      __syncthreads();

      // output projection: 16 tiles
      gemm_tile_step(p.att16, p.sawo + (size_t)l * 65536, p.sa_bo + l * 256,
                     256, GM_PLAIN, t, 256, w, lane,
                     p.proj, nullptr, nullptr, nullptr, nullptr, nullptr, nullptr);
      __syncthreads();
      if (w < 8) ln_wave(w, lane, p.x32, p.proj, p.ln1_g + l * 256, p.ln1_b + l * 256, p.x16);
      __syncthreads();

      // cross-attn Q projection: 16 tiles
      gemm_tile_step(p.x16, p.cawq + (size_t)l * 65536, p.ca_bqkv + l * 768,
                     256, GM_PLAIN, t, 256, w, lane,
                     p.q32, nullptr, nullptr, nullptr, nullptr, nullptr, nullptr);
      __syncthreads();
      for (int task = w; task < 64; task += 16)
        cross_attn_task(p, l, task, lane, ssw, sqw);
      __syncthreads();
      gemm_tile_step(p.att16, p.cawo + (size_t)l * 65536, p.ca_bo + l * 256,
                     256, GM_PLAIN, t, 256, w, lane,
                     p.proj, nullptr, nullptr, nullptr, nullptr, nullptr, nullptr);
      __syncthreads();
      if (w < 8) ln_wave(w, lane, p.x32, p.proj, p.ln2_g + l * 256, p.ln2_b + l * 256, p.x16);
      __syncthreads();

      // FFN: 64 tiles (4/wave) then 16 tiles
      for (int tile = w; tile < 64; tile += 16)
        gemm_tile_step(p.x16, p.ffw1 + (size_t)l * 262144, p.ffn_b1 + l * 1024,
                       256, GM_RELU, t, 1024, tile, lane,
                       nullptr, p.ffn16, nullptr, nullptr, nullptr, nullptr, nullptr);
      __syncthreads();
      gemm_tile_step(p.ffn16, p.ffw2 + (size_t)l * 262144, p.ffn_b2 + l * 256,
                     1024, GM_PLAIN, t, 256, w, lane,
                     p.proj, nullptr, nullptr, nullptr, nullptr, nullptr, nullptr);
      __syncthreads();
      if (w < 8) ln_wave(w, lane, p.x32, p.proj, p.ln3_g + l * 256, p.ln3_b + l * 256, p.x16);
      __syncthreads();
    }

    // output head: linear -> BN affine -> relu -> linear
    gemm_tile_step(p.x16, p.opw1, p.op_b1, 256, GM_HEADBN, t, 256, w, lane,
                   nullptr, p.head16, nullptr, nullptr, p.bn_g, p.bn_b, nullptr);
    __syncthreads();
    // 8 tiles over 16 waves: duplicate (tile = w&7) so EXEC stays full around WMMA;
    // duplicated waves write identical values.
    gemm_tile_step(p.head16, p.opw2, p.op_b2, 256, GM_OUT, t, 128, w & 7, lane,
                   nullptr, p.cur16, nullptr, nullptr, nullptr, nullptr, p.dout);
    __syncthreads();
  }
}

// ---------------- host launch ----------------
extern "C" void kernel_launch(void* const* d_in, const int* in_sizes, int n_in,
                              void* d_out, int out_size, void* d_ws, size_t ws_size,
                              hipStream_t stream) {
  const float* enc     = (const float*)d_in[0];
  const float* emb_w   = (const float*)d_in[1];
  const float* emb_b   = (const float*)d_in[2];
  const float* sa_wqkv = (const float*)d_in[3];
  const float* sa_bqkv = (const float*)d_in[4];
  const float* sa_wo   = (const float*)d_in[5];
  const float* sa_bo   = (const float*)d_in[6];
  const float* ca_wqkv = (const float*)d_in[7];
  const float* ca_bqkv = (const float*)d_in[8];
  const float* ca_wo   = (const float*)d_in[9];
  const float* ca_bo   = (const float*)d_in[10];
  const float* ln1_g   = (const float*)d_in[11];
  const float* ln1_b   = (const float*)d_in[12];
  const float* ln2_g   = (const float*)d_in[13];
  const float* ln2_b   = (const float*)d_in[14];
  const float* ffn_w1  = (const float*)d_in[15];
  const float* ffn_b1  = (const float*)d_in[16];
  const float* ffn_w2  = (const float*)d_in[17];
  const float* ffn_b2  = (const float*)d_in[18];
  const float* ln3_g   = (const float*)d_in[19];
  const float* ln3_b   = (const float*)d_in[20];
  const float* op_w1   = (const float*)d_in[21];
  const float* op_b1   = (const float*)d_in[22];
  const float* bn_g    = (const float*)d_in[23];
  const float* bn_b    = (const float*)d_in[24];
  const float* op_w2   = (const float*)d_in[25];
  const float* op_b2   = (const float*)d_in[26];
  (void)in_sizes; (void)n_in; (void)out_size; (void)ws_size;

  float*     dout = (float*)d_out;
  _Float16*  h    = (_Float16*)d_ws;
  float*     f    = (float*)((char*)d_ws + F32_BYTE_OFF);

  auto cvt = [&](const float* s, _Float16* d, int n) {
    cvt_f32_f16<<<dim3((n + 255) / 256), dim3(256), 0, stream>>>(s, d, n);
  };

  // one-time f16 shadows
  cvt(emb_w,             h + OFF_EMBW,           32768);
  cvt(sa_wqkv,           h + OFF_SAQKV,          393216);
  cvt(sa_wo,             h + OFF_SAWO,           131072);
  cvt(ca_wqkv,           h + OFF_CAWQ,           65536);    // layer0 Wq
  cvt(ca_wqkv + 196608,  h + OFF_CAWQ + 65536,   65536);    // layer1 Wq
  cvt(ca_wqkv + 65536,   h + OFF_CAWKV,          131072);   // layer0 Wkv
  cvt(ca_wqkv + 262144,  h + OFF_CAWKV + 131072, 131072);   // layer1 Wkv
  cvt(ca_wo,             h + OFF_CAWO,           131072);
  cvt(ffn_w1,            h + OFF_FFW1,           524288);
  cvt(ffn_w2,            h + OFF_FFW2,           524288);
  cvt(op_w1,             h + OFF_OPW1,           65536);
  cvt(op_w2,             h + OFF_OPW2,           32768);
  cvt(enc,               h + OFF_ENC,            524288);
  zero_f16<<<dim3((30720 + 255) / 256), dim3(256), 0, stream>>>(h + OFF_X16, 30720);

  // one-time cross-attn K/V (full-GPU WMMA GEMM)
  memkv_gemm<<<dim3(8192), dim3(32), 0, stream>>>(h + OFF_ENC, h + OFF_CAWKV,
                                                  ca_bqkv, f + OFF_MEMK, f + OFF_MEMV);

  // the whole 64-step serial decode: ONE persistent launch
  DecodeParams p;
  p.embw = h + OFF_EMBW;  p.saqkv = h + OFF_SAQKV;  p.sawo = h + OFF_SAWO;
  p.cawq = h + OFF_CAWQ;  p.cawo  = h + OFF_CAWO;
  p.ffw1 = h + OFF_FFW1;  p.ffw2  = h + OFF_FFW2;
  p.opw1 = h + OFF_OPW1;  p.opw2  = h + OFF_OPW2;
  p.x16 = h + OFF_X16;    p.att16 = h + OFF_ATT16; p.ffn16 = h + OFF_FFN16;
  p.head16 = h + OFF_HEAD16; p.cur16 = h + OFF_CUR16;
  p.emb_b = emb_b; p.sa_bqkv = sa_bqkv; p.sa_bo = sa_bo;
  p.ca_bqkv = ca_bqkv; p.ca_bo = ca_bo;
  p.ln1_g = ln1_g; p.ln1_b = ln1_b; p.ln2_g = ln2_g; p.ln2_b = ln2_b;
  p.ln3_g = ln3_g; p.ln3_b = ln3_b;
  p.ffn_b1 = ffn_b1; p.ffn_b2 = ffn_b2;
  p.op_b1 = op_b1; p.bn_g = bn_g; p.bn_b = bn_b; p.op_b2 = op_b2;
  p.mem_k = f + OFF_MEMK; p.mem_v = f + OFF_MEMV;
  p.kc = f + OFF_KC; p.vc = f + OFF_VC;
  p.x32 = f + OFF_X32; p.q32 = f + OFF_Q32; p.proj = f + OFF_PROJ;
  p.dout = dout;
  decode_persistent<<<dim3(1), dim3(512), 0, stream>>>(p);
}
